// Attention_49271864820258
// MI455X (gfx1250) — matrix operand order repaired and checked by reading, not compiled, not run on hardware
//
#include <hip/hip_runtime.h>
#include <hip/hip_bf16.h>

// ---- types ----
typedef __attribute__((ext_vector_type(8)))  __bf16 v8bf;
typedef __attribute__((ext_vector_type(16))) __bf16 v16bf;
typedef __attribute__((ext_vector_type(8)))  float  v8f;
typedef __attribute__((ext_vector_type(8)))  float  f32x8;
typedef __attribute__((ext_vector_type(4)))  unsigned int u32x4;
typedef __attribute__((ext_vector_type(8)))  int    i32x8;
typedef __attribute__((ext_vector_type(4)))  int    i32x4;

#define HIDDEN 1024
#define NHEADS 16
#define HDIM   64
#define SEQ    2048
#define BATCH  2

#if defined(__AMDGCN__) && __has_builtin(__builtin_amdgcn_tensor_load_to_lds)
#define HAVE_TDM 1
#else
#define HAVE_TDM 0
#endif

// ---- WMMA helper: D = A(16x32 bf16) * B(32x16 bf16) + C(16x16 f32) ----
__device__ inline v8f wmma_bf16(v16bf a, v16bf b, v8f c) {
  return __builtin_amdgcn_wmma_f32_16x16x32_bf16(
      /*neg_a=*/false, a, /*neg_b=*/false, b,
      /*c_mod=*/(short)0, c, /*reuse_a=*/false, /*reuse_b=*/false);
}

__device__ inline v16bf cat8(v8bf lo, v8bf hi) {
  return __builtin_shufflevector(lo, hi, 0,1,2,3,4,5,6,7,8,9,10,11,12,13,14,15);
}

// A-operand (16x32 bf16), contiguous k order (B loads match): lane = (m = l&15, h = l>>4),
// elems 0..15 = K[16h..16h+16). p -> (row m, chunk+16h).
__device__ inline v16bf load_a_bf16(const __bf16* p) {
  v8bf lo = *(const v8bf*)p;
  v8bf hi = *(const v8bf*)(p + 8);
  return cat8(lo, hi);
}

__device__ inline v16bf load_a_f32(const float* p) {
  f32x8 lo = *(const f32x8*)p;
  f32x8 hi = *(const f32x8*)(p + 8);
  v16bf r;
#pragma unroll
  for (int i = 0; i < 8; ++i) { r[i] = (__bf16)lo[i]; r[i + 8] = (__bf16)hi[i]; }
  return r;
}

#if HAVE_TDM
// ---- Tensor Data Mover: 2-D bf16 tile (tile_d0 contiguous elems x tile_d1 rows,
// row stride stride0 elems) from global -> LDS at byte offset lds_off.
// D# per CDNA5 ISA: group0 = {flags, lds_addr, global_addr, type=2},
// group1 = {data_size=2B, tensor dims = tile dims (no OOB clip), tile dims, stride}.
__device__ __forceinline__ void tdm_load_2d(unsigned lds_off, const void* gptr,
                                            unsigned tile_d0, unsigned tile_d1,
                                            unsigned stride0) {
  unsigned long long ga = (unsigned long long)gptr;
  u32x4 g0;
  g0[0] = 1u;                                   // count=1, user descriptor
  g0[1] = lds_off;                              // lds_addr (bytes)
  g0[2] = (unsigned)ga;                         // global_addr[31:0]
  g0[3] = (unsigned)(ga >> 32) | (2u << 30);    // global_addr[56:32] | type=2
  i32x8 g1;
  g1[0] = (int)(1u << 16);                      // wg_mask=0, data_size=1 (2 bytes)
  g1[1] = (int)((tile_d0 & 0xFFFFu) << 16);     // tensor_dim0[15:0] (= tile_d0)
  g1[2] = (int)((tile_d0 >> 16) | ((tile_d1 & 0xFFFFu) << 16));  // dim0 hi | dim1 lo
  g1[3] = (int)((tile_d1 >> 16) | ((tile_d0 & 0xFFFFu) << 16));  // dim1 hi | tile_dim0
  g1[4] = (int)(tile_d1 & 0xFFFFu);             // tile_dim1; tile_dim2=0
  g1[5] = (int)stride0;                         // tensor_dim0_stride[31:0]
  g1[6] = 0;                                    // stride0[47:32] | stride1[15:0]
  g1[7] = 0;
  i32x4 gz = {};
#if __has_include(<hip/amd_detail/amd_gfx1250_TDM.h>)
  i32x8 gz8 = {};
  __builtin_amdgcn_tensor_load_to_lds(g0, g1, gz, gz, gz8, 0);   // 6-arg toolchain
#else
  __builtin_amdgcn_tensor_load_to_lds(g0, g1, gz, gz, 0);        // 5-arg (ROCm 7.2)
#endif
}
#endif  // HAVE_TDM

// ---- weight transpose + f32->bf16 convert: dst[o*1024+k] = src[k*1024+o]*scale ----
__global__ void wtrans_kernel(const float* __restrict__ src,
                              __bf16* __restrict__ dst, float scale) {
  int idx = blockIdx.x * blockDim.x + threadIdx.x;   // 1M elements
  int o = idx >> 10;
  int k = idx & 1023;
  dst[idx] = (__bf16)(src[k * HIDDEN + o] * scale);
}

// ---- GEMM: out[M x 1024] = A(f32, M x 1024) * Bt^T (Bt = [1024 out][1024 K] bf16) ----
// grid (M/128, 16), block 128 (4 waves); wave = 32(M) x 64(N). B panel staged to LDS
// by TDM, double-buffered, one DMA per workgroup per K step (kills 4x-redundant VMEM).
__global__ void __launch_bounds__(128)
gemm_bf16_kernel(const float* __restrict__ A, const __bf16* __restrict__ Bt,
                 float* __restrict__ outF, __bf16* __restrict__ outB, int mode) {
  const int lane = threadIdx.x & 31;
  const int wave = threadIdx.x >> 5;
  const int n16 = lane & 15;
  const int h   = lane >> 4;
  const int mwave = blockIdx.x * 128 + wave * 32;
  const int nblk  = blockIdx.y * 64;

  v8f acc[2][4] = {};
  const float* a0p = A + (size_t)(mwave + n16) * HIDDEN + 16 * h;
  const float* a1p = A + (size_t)(mwave + 16 + n16) * HIDDEN + 16 * h;

#if HAVE_TDM
  __shared__ __align__(16) __bf16 bstage[2][64 * 32];   // [out-row][K] tiles, 4KB each
  const unsigned lds0 = (unsigned)(unsigned long long)&bstage[0][0];
  const unsigned lds1 = (unsigned)(unsigned long long)&bstage[1][0];
  const __bf16* bpanel = Bt + (size_t)nblk * HIDDEN;
  if (wave == 0)
    tdm_load_2d(lds0, bpanel, /*d0=*/32, /*d1=*/64, /*stride=*/HIDDEN);
  for (int kc = 0; kc < HIDDEN / 32; ++kc) {
    const int k0 = kc * 32;
    if (wave == 0) {
      if (kc + 1 < HIDDEN / 32) {
        tdm_load_2d((kc & 1) ? lds0 : lds1, bpanel + k0 + 32, 32, 64, HIDDEN);
        __builtin_amdgcn_s_wait_tensorcnt((short)1);   // current buffer complete
      } else {
        __builtin_amdgcn_s_wait_tensorcnt((short)0);
      }
    }
    __syncthreads();                                   // buffer ready for all waves
    __builtin_prefetch(a0p + k0 + 64, 0, 1);           // global_prefetch_b8
    v16bf a0 = load_a_f32(a0p + k0);
    v16bf a1 = load_a_f32(a1p + k0);
    const __bf16* bl = (kc & 1) ? &bstage[1][0] : &bstage[0][0];
    v16bf bt[4];
#pragma unroll
    for (int j = 0; j < 4; ++j)
      bt[j] = *(const v16bf*)(bl + (16 * j + n16) * 32 + 16 * h);
#pragma unroll
    for (int j = 0; j < 4; ++j) {
      acc[0][j] = wmma_bf16(a0, bt[j], acc[0][j]);
      acc[1][j] = wmma_bf16(a1, bt[j], acc[1][j]);
    }
    __syncthreads();                                   // done consuming; next DMA may overwrite
  }
#else
  for (int k0 = 0; k0 < HIDDEN; k0 += 32) {
    v16bf a0 = load_a_f32(a0p + k0);
    v16bf a1 = load_a_f32(a1p + k0);
    v16bf bt[4];
#pragma unroll
    for (int j = 0; j < 4; ++j)
      bt[j] = *(const v16bf*)(Bt + (size_t)(nblk + 16 * j + n16) * HIDDEN + k0 + 16 * h);
#pragma unroll
    for (int j = 0; j < 4; ++j) {
      acc[0][j] = wmma_bf16(a0, bt[j], acc[0][j]);
      acc[1][j] = wmma_bf16(a1, bt[j], acc[1][j]);
    }
  }
#endif

#pragma unroll
  for (int t = 0; t < 2; ++t) {
    const int mb = mwave + t * 16 + 8 * h;   // C layout: elem i -> row mb+i
#pragma unroll
    for (int j = 0; j < 4; ++j) {
      const int nb = nblk + 16 * j + n16;
#pragma unroll
      for (int i = 0; i < 8; ++i) {
        const int m = mb + i;
        const float val = acc[t][j][i];
        if (mode == 3) {
          outF[(size_t)m * HIDDEN + nb] = val;
        } else {
          const int bb = m >> 11;      // batch
          const int s  = m & 2047;     // seq position
          const int hd = nb >> 6;      // head
          const int hh = nb & 63;      // dim-in-head
          size_t idx;
          if (mode == 2)  // V^T: [B, N, H, T]
            idx = ((size_t)(bb * NHEADS + hd) * HDIM + hh) * SEQ + s;
          else            // Q/K: [B, N, S, H]
            idx = ((size_t)(bb * NHEADS + hd) * SEQ + s) * HDIM + hh;
          outB[idx] = (__bf16)val;
        }
      }
    }
  }
}

// ---- fused flash attention: softmax(Q K^T + bias) V, per (b, head, 64 F rows) ----
// grid (F/64=32, NHEADS=16, B=2), block 128 (4 waves, each owns 16 F rows).
// K (32x64) and V^T (64x32) tiles staged to LDS by TDM, double-buffered.
__global__ void __launch_bounds__(128)
flash_attn_kernel(const __bf16* __restrict__ Qh, const __bf16* __restrict__ Kh,
                  const __bf16* __restrict__ Vt, const float* __restrict__ bias,
                  float* __restrict__ attnOut) {
  __shared__ __align__(16) __bf16 plds[4][16 * 32];  // per-wave P tile [16 f][32 t]
  const int lane = threadIdx.x & 31;
  const int wave = threadIdx.x >> 5;
  const int n16 = lane & 15;
  const int h   = lane >> 4;
  const int b    = blockIdx.z;
  const int head = blockIdx.y;
  const int fbase = blockIdx.x * 64 + wave * 16;

  const __bf16* q  = Qh + (size_t)(b * NHEADS + head) * SEQ * HDIM;
  const __bf16* km = Kh + (size_t)(b * NHEADS + head) * SEQ * HDIM;
  const __bf16* vt = Vt + (size_t)(b * NHEADS + head) * HDIM * SEQ;
  const float*  bb = bias + (size_t)b * SEQ * SEQ + (size_t)(fbase + 8 * h) * SEQ;

  // Q tile A-operands, resident for whole T loop (K-dim = H = 64 -> 2 chunks)
  v16bf aq0 = load_a_bf16(q + (size_t)(fbase + n16) * HDIM + 16 * h);
  v16bf aq1 = load_a_bf16(q + (size_t)(fbase + n16) * HDIM + 32 + 16 * h);

  v8f o0 = {}, o1 = {}, o2 = {}, o3 = {};
  float mx[8], sm[8];
#pragma unroll
  for (int i = 0; i < 8; ++i) { mx[i] = -1e30f; sm[i] = 0.0f; }
  __bf16* pw = plds[wave];

#if HAVE_TDM
  __shared__ __align__(16) __bf16 kstage[2][32 * 64];  // [t-row][h] tiles, 4KB
  __shared__ __align__(16) __bf16 vstage[2][64 * 32];  // [h-row][t] tiles, 4KB
  unsigned klds[2], vlds[2];
  klds[0] = (unsigned)(unsigned long long)&kstage[0][0];
  klds[1] = (unsigned)(unsigned long long)&kstage[1][0];
  vlds[0] = (unsigned)(unsigned long long)&vstage[0][0];
  vlds[1] = (unsigned)(unsigned long long)&vstage[1][0];
  if (wave == 0) {
    tdm_load_2d(klds[0], km, /*d0=*/64, /*d1=*/32, /*stride=*/HDIM);
    tdm_load_2d(vlds[0], vt, /*d0=*/32, /*d1=*/64, /*stride=*/SEQ);
  }
#endif

  for (int t0 = 0; t0 < SEQ; t0 += 32) {
    v8f cl = {}, ch = {};
#if HAVE_TDM
    const int buf = (t0 >> 5) & 1;
    if (wave == 0) {
      if (t0 + 32 < SEQ) {
        tdm_load_2d(klds[buf ^ 1], km + (size_t)(t0 + 32) * HDIM, 64, 32, HDIM);
        tdm_load_2d(vlds[buf ^ 1], vt + (t0 + 32), 32, 64, SEQ);
        __builtin_amdgcn_s_wait_tensorcnt((short)2);   // current K/V pair complete
      } else {
        __builtin_amdgcn_s_wait_tensorcnt((short)0);
      }
    }
    __syncthreads();                                   // tiles visible to all waves
    {
      const __bf16* kl = kstage[buf];
      v16bf b0 = *(const v16bf*)(kl + (size_t)n16 * HDIM + 16 * h);
      v16bf b1 = *(const v16bf*)(kl + (size_t)n16 * HDIM + 32 + 16 * h);
      cl = wmma_bf16(aq0, b0, cl);
      cl = wmma_bf16(aq1, b1, cl);
      v16bf b2 = *(const v16bf*)(kl + (size_t)(16 + n16) * HDIM + 16 * h);
      v16bf b3 = *(const v16bf*)(kl + (size_t)(16 + n16) * HDIM + 32 + 16 * h);
      ch = wmma_bf16(aq0, b2, ch);
      ch = wmma_bf16(aq1, b3, ch);
    }
#else
    {
      v16bf b0 = *(const v16bf*)(km + (size_t)(t0 + n16) * HDIM + 16 * h);
      v16bf b1 = *(const v16bf*)(km + (size_t)(t0 + n16) * HDIM + 32 + 16 * h);
      cl = wmma_bf16(aq0, b0, cl);
      cl = wmma_bf16(aq1, b1, cl);
      v16bf b2 = *(const v16bf*)(km + (size_t)(t0 + 16 + n16) * HDIM + 16 * h);
      v16bf b3 = *(const v16bf*)(km + (size_t)(t0 + 16 + n16) * HDIM + 32 + 16 * h);
      ch = wmma_bf16(aq0, b2, ch);
      ch = wmma_bf16(aq1, b3, ch);
    }
#endif
    // bias add (C layout: elem i -> row fbase+8h+i, col n16 / 16+n16)
    const float* br = bb + t0;
#pragma unroll
    for (int i = 0; i < 8; ++i) {
      cl[i] += br[(size_t)i * SEQ + n16];
      ch[i] += br[(size_t)i * SEQ + 16 + n16];
    }
    // online softmax; row stats reduced across the owning 16-lane group
#pragma unroll
    for (int i = 0; i < 8; ++i) {
      float cmax = fmaxf(cl[i], ch[i]);
      cmax = fmaxf(cmax, __shfl_xor(cmax, 1, 16));
      cmax = fmaxf(cmax, __shfl_xor(cmax, 2, 16));
      cmax = fmaxf(cmax, __shfl_xor(cmax, 4, 16));
      cmax = fmaxf(cmax, __shfl_xor(cmax, 8, 16));
      float mnew = fmaxf(mx[i], cmax);
      float corr = __expf(mx[i] - mnew);
      float pl = __expf(cl[i] - mnew);
      float ph = __expf(ch[i] - mnew);
      float rs = pl + ph;
      rs += __shfl_xor(rs, 1, 16);
      rs += __shfl_xor(rs, 2, 16);
      rs += __shfl_xor(rs, 4, 16);
      rs += __shfl_xor(rs, 8, 16);
      sm[i] = sm[i] * corr + rs;
      mx[i] = mnew;
      o0[i] *= corr; o1[i] *= corr; o2[i] *= corr; o3[i] *= corr;
      const int r = 8 * h + i;                      // stage P (C layout -> [16][32])
      pw[r * 32 + n16]      = (__bf16)pl;
      pw[r * 32 + 16 + n16] = (__bf16)ph;
    }
    __syncthreads();
    // re-load P in A-operand layout (K-dim = 32 t values)
    v16bf pa = load_a_bf16(pw + n16 * 32 + 16 * h);
#if HAVE_TDM
    {
      const __bf16* vl = vstage[buf];
      v16bf bv0 = *(const v16bf*)(vl + (size_t)(n16)      * 32 + 16 * h);
      v16bf bv1 = *(const v16bf*)(vl + (size_t)(16 + n16) * 32 + 16 * h);
      v16bf bv2 = *(const v16bf*)(vl + (size_t)(32 + n16) * 32 + 16 * h);
      v16bf bv3 = *(const v16bf*)(vl + (size_t)(48 + n16) * 32 + 16 * h);
      o0 = wmma_bf16(pa, bv0, o0);
      o1 = wmma_bf16(pa, bv1, o1);
      o2 = wmma_bf16(pa, bv2, o2);
      o3 = wmma_bf16(pa, bv3, o3);
    }
#else
    {
      v16bf bv0 = *(const v16bf*)(vt + (size_t)(n16)      * SEQ + t0 + 16 * h);
      v16bf bv1 = *(const v16bf*)(vt + (size_t)(16 + n16) * SEQ + t0 + 16 * h);
      v16bf bv2 = *(const v16bf*)(vt + (size_t)(32 + n16) * SEQ + t0 + 16 * h);
      v16bf bv3 = *(const v16bf*)(vt + (size_t)(48 + n16) * SEQ + t0 + 16 * h);
      o0 = wmma_bf16(pa, bv0, o0);
      o1 = wmma_bf16(pa, bv1, o1);
      o2 = wmma_bf16(pa, bv2, o2);
      o3 = wmma_bf16(pa, bv3, o3);
    }
#endif
    __syncthreads();   // consumption done; next iteration's DMA/P writes are safe
  }

  // normalize and write attn [B, F, N*H] f32 (A matrix of the final projection)
  const int fr = fbase + 8 * h;
  const int colb = head * HDIM;
#pragma unroll
  for (int i = 0; i < 8; ++i) {
    const float inv = 1.0f / sm[i];
    const size_t base = ((size_t)b * SEQ + (fr + i)) * HIDDEN + colb;
    attnOut[base + n16]      = o0[i] * inv;
    attnOut[base + 16 + n16] = o1[i] * inv;
    attnOut[base + 32 + n16] = o2[i] * inv;
    attnOut[base + 48 + n16] = o3[i] * inv;
  }
}

extern "C" void kernel_launch(void* const* d_in, const int* in_sizes, int n_in,
                              void* d_out, int out_size, void* d_ws, size_t ws_size,
                              hipStream_t stream) {
  const float* qin  = (const float*)d_in[0];  // [B,F,1024]
  const float* sin_ = (const float*)d_in[1];  // [B,T,1024]
  const float* bias = (const float*)d_in[2];  // [B,1,F,T]
  const float* wq   = (const float*)d_in[3];  // [1024,16,64]
  const float* wk   = (const float*)d_in[4];
  const float* wv   = (const float*)d_in[5];
  const float* wo   = (const float*)d_in[6];  // [16,64,1024]
  float* out = (float*)d_out;

  char* ws = (char*)d_ws;
  __bf16* wqT = (__bf16*)(ws + (0ull  << 20));  // 2 MB each
  __bf16* wkT = (__bf16*)(ws + (2ull  << 20));
  __bf16* wvT = (__bf16*)(ws + (4ull  << 20));
  __bf16* woT = (__bf16*)(ws + (6ull  << 20));
  __bf16* Qh  = (__bf16*)(ws + (8ull  << 20));  // [B,N,F,H] bf16, 8 MB
  __bf16* Kh  = (__bf16*)(ws + (16ull << 20));  // [B,N,T,H] bf16, 8 MB
  __bf16* Vt  = (__bf16*)(ws + (24ull << 20));  // [B,N,H,T] bf16, 8 MB
  float*  att = (float*)(ws + (32ull << 20));   // [B,F,1024] f32, 16 MB

  // weights -> bf16 [out][K]; fold HEAD_DIM^-0.5 = 0.125 into wq
  wtrans_kernel<<<4096, 256, 0, stream>>>(wq, wqT, 0.125f);
  wtrans_kernel<<<4096, 256, 0, stream>>>(wk, wkT, 1.0f);
  wtrans_kernel<<<4096, 256, 0, stream>>>(wv, wvT, 1.0f);
  wtrans_kernel<<<4096, 256, 0, stream>>>(wo, woT, 1.0f);

  dim3 gg(32, 16);   // M=4096 / 128, 1024 / 64
  dim3 bk(128);
  gemm_bf16_kernel<<<gg, bk, 0, stream>>>(qin,  wqT, nullptr, Qh, 0);  // Q (scaled)
  gemm_bf16_kernel<<<gg, bk, 0, stream>>>(sin_, wkT, nullptr, Kh, 0);  // K
  gemm_bf16_kernel<<<gg, bk, 0, stream>>>(sin_, wvT, nullptr, Vt, 2);  // V (transposed)

  flash_attn_kernel<<<dim3(32, 16, 2), bk, 0, stream>>>(Qh, Kh, Vt, bias, att);

  gemm_bf16_kernel<<<gg, bk, 0, stream>>>(att, woT, out, nullptr, 3);  // out proj
}
